// MultiHeadAttention_81149112090726
// MI455X (gfx1250) — matrix operand order repaired and checked by reading, not compiled, not run on hardware
//
#include <hip/hip_runtime.h>
#include <hip/hip_bf16.h>
#include <math.h>

typedef __attribute__((ext_vector_type(16))) _Float16 v16h;
typedef __attribute__((ext_vector_type(8)))  _Float16 v8h;
typedef __attribute__((ext_vector_type(8)))  float    v8f;
typedef __attribute__((ext_vector_type(4)))  float    v4f;

#define WMMA_F16(A, B, C) \
  __builtin_amdgcn_wmma_f32_16x16x32_f16(false, (A), false, (B), (short)0, (C), false, false)

// WMMA 16-bit A-matrix layout (ISA 7.12.2): per lane the 16 halves are two
// contiguous 8-element K chunks:  [kBase .. kBase+7] and [16+kBase .. 16+kBase+7]
// with kBase = 8*(lane>=16).
// B-matrix layout: same K order as A, N = lane%16; kernels 2/3 use K = 16*(lane>=16) + j in both A and B.
// C/D layout: row M = r + 8*(lane>=16), col N = lane%16.

// ---------------------------------------------------------------------------
// Kernel 1: QKV projection GEMM.  out[b,o,n] = sum_i W[o,i]*X[b,i,n] + bias[o]
// Output f16 in attention-friendly layouts (o = d*4 + h):
//   mode 0/1 (Q,K): ws[((b*4+h)*2048 + n)*64 + d]   (row n, contiguous d)
//   mode 2   (V)  : ws[((b*4+h)*64 + d)*2048 + n]   (row d, contiguous n)
// One wave computes a 16(o) x 64(n) tile: 32 WMMAs.
// ---------------------------------------------------------------------------
__global__ __launch_bounds__(128) void proj_qkv_kernel(
    const float* __restrict__ X, const float* __restrict__ W,
    const float* __restrict__ bias, _Float16* __restrict__ out, int mode) {
  const int lane   = threadIdx.x & 31;
  const int wave   = threadIdx.x >> 5;
  const int laneHi = lane >> 4;
  const int lmod   = lane & 15;
  const int kBase  = laneHi * 8;

  const int b     = blockIdx.z;
  const int oBase = blockIdx.y * 16;
  const int nBase = (blockIdx.x * 4 + wave) * 64;

  const float* Wrow = W + (size_t)(oBase + lmod) * 256;
  const float* Xb   = X + (size_t)b * 256 * 2048;

  v8f acc[4] = {v8f{0}, v8f{0}, v8f{0}, v8f{0}};

  for (int k0 = 0; k0 < 256; k0 += 32) {
    // A tile: weights, two contiguous 8-float chunks -> 4x float4 loads
    v4f w0 = *(const v4f*)(Wrow + k0 + kBase);
    v4f w1 = *(const v4f*)(Wrow + k0 + kBase + 4);
    v4f w2 = *(const v4f*)(Wrow + k0 + 16 + kBase);
    v4f w3 = *(const v4f*)(Wrow + k0 + 16 + kBase + 4);
    v16h a;
#pragma unroll
    for (int j = 0; j < 4; ++j) {
      a[j]      = (_Float16)w0[j];
      a[4 + j]  = (_Float16)w1[j];
      a[8 + j]  = (_Float16)w2[j];
      a[12 + j] = (_Float16)w3[j];
    }
#pragma unroll
    for (int t = 0; t < 4; ++t) {
      const int n = nBase + t * 16 + lmod;
      v16h bm;
#pragma unroll
      for (int j = 0; j < 16; ++j) {
        const int i = k0 + (j < 8 ? kBase + j : 16 + kBase + (j - 8));  // B-layout: K = kBase + j (j<8), 16 + kBase + j-8 (j>=8)
        bm[j] = (_Float16)Xb[(size_t)i * 2048 + n];
      }
      acc[t] = WMMA_F16(a, bm, acc[t]);
    }
  }

  // Epilogue: bias, convert, scatter into head-split layout
#pragma unroll
  for (int t = 0; t < 4; ++t) {
    const int n = nBase + t * 16 + lmod;
#pragma unroll
    for (int r = 0; r < 8; ++r) {
      const int o = oBase + r + laneHi * 8;  // C-layout row
      const _Float16 val = (_Float16)(acc[t][r] + bias[o]);
      const int d = o >> 2, h = o & 3;
      if (mode == 2)
        out[(((size_t)b * 4 + h) * 64 + d) * 2048 + n] = val;
      else
        out[(((size_t)b * 4 + h) * 2048 + n) * 64 + d] = val;
    }
  }
}

// ---------------------------------------------------------------------------
// Kernel 2: flash attention per (b, h).  One wave owns 16 query rows.
// 64-key steps: 8 WMMAs for S, one online-softmax pass, 8 WMMAs for P*V.
// Result stored channel-contiguous: xw[b][n][c] with c = d*4 + h.
// ---------------------------------------------------------------------------
__global__ __launch_bounds__(128) void attn_kernel(
    const _Float16* __restrict__ qw, const _Float16* __restrict__ kw,
    const _Float16* __restrict__ vw, _Float16* __restrict__ xw) {
  __shared__ __align__(32) _Float16 lds[4][16 * 64];  // per-wave P bounce (C->A relayout)

  const int lane   = threadIdx.x & 31;
  const int wave   = threadIdx.x >> 5;
  const int laneHi = lane >> 4;
  const int lmod   = lane & 15;
  const int kBase  = laneHi * 8;

  const int b = blockIdx.z, h = blockIdx.y;
  const int nBase = (blockIdx.x * 4 + wave) * 16;
  const size_t bh = (size_t)b * 4 + h;

  const _Float16* Q = qw + bh * 2048 * 64;
  const _Float16* K = kw + bh * 2048 * 64;
  const _Float16* V = vw + bh * 64 * 2048;

  // Q A-tiles (d 0..31 and d 32..63): 4x 16B loads each, loaded once
  const _Float16* Qrow = Q + (size_t)(nBase + lmod) * 64;
  v16h aq0, aq1;
  {
    v8h c0 = *(const v8h*)(Qrow + laneHi * 16);
    v8h c1 = *(const v8h*)(Qrow + laneHi * 16 + 8);
    v8h c2 = *(const v8h*)(Qrow + 32 + laneHi * 16);
    v8h c3 = *(const v8h*)(Qrow + 40 + laneHi * 16);
#pragma unroll
    for (int j = 0; j < 8; ++j) {
      aq0[j] = c0[j]; aq0[8 + j] = c1[j];
      aq1[j] = c2[j]; aq1[8 + j] = c3[j];
    }
  }

  v8f accO[4] = {v8f{0}, v8f{0}, v8f{0}, v8f{0}};
  float rowMax[8], rowSum[8];
#pragma unroll
  for (int r = 0; r < 8; ++r) { rowMax[r] = -3.0e38f; rowSum[r] = 0.0f; }

  const float scale = 0.125f;  // 1/sqrt(64)
  _Float16* myLds = &lds[wave][0];

  for (int m0 = 0; m0 < 2048; m0 += 64) {
    // Prefetch next 64-key block of K and V (global_prefetch_b8)
    if (m0 + 64 < 2048) {
      __builtin_prefetch(K + (size_t)(m0 + 64 + lane) * 64, 0, 1);
      __builtin_prefetch(K + (size_t)(m0 + 96 + lane) * 64, 0, 1);
      __builtin_prefetch(V + (size_t)lane * 2048 + m0 + 64, 0, 1);
      __builtin_prefetch(V + (size_t)(32 + lane) * 2048 + m0 + 64, 0, 1);
    }

    // ---- S = Q Kt : four 16x16 subtiles, K-dim 64 = 2 chained WMMAs each ----
    v8f s[4] = {v8f{0}, v8f{0}, v8f{0}, v8f{0}};
#pragma unroll
    for (int u = 0; u < 4; ++u) {
      const _Float16* Kr = K + (size_t)(m0 + u * 16 + lmod) * 64 + laneHi * 16;
      v16h bk0 = *(const v16h*)(Kr);       // d 0..31
      v16h bk1 = *(const v16h*)(Kr + 32);  // d 32..63
      s[u] = WMMA_F16(aq0, bk0, s[u]);
      s[u] = WMMA_F16(aq1, bk1, s[u]);
    }

    // ---- online softmax (rows live in 16-lane halves; xor-reduce 1/2/4/8) ---
#pragma unroll
    for (int r = 0; r < 8; ++r) {
      const float v0 = s[0][r] * scale;
      const float v1 = s[1][r] * scale;
      const float v2 = s[2][r] * scale;
      const float v3 = s[3][r] * scale;
      float mx = fmaxf(fmaxf(v0, v1), fmaxf(v2, v3));
      mx = fmaxf(mx, __shfl_xor(mx, 1, 32));
      mx = fmaxf(mx, __shfl_xor(mx, 2, 32));
      mx = fmaxf(mx, __shfl_xor(mx, 4, 32));
      mx = fmaxf(mx, __shfl_xor(mx, 8, 32));
      const float newMax = fmaxf(rowMax[r], mx);
      const float corr   = __expf(rowMax[r] - newMax);
      rowMax[r] = newMax;
      const float p0 = __expf(v0 - newMax);
      const float p1 = __expf(v1 - newMax);
      const float p2 = __expf(v2 - newMax);
      const float p3 = __expf(v3 - newMax);
      float ps = (p0 + p1) + (p2 + p3);
      ps += __shfl_xor(ps, 1, 32);
      ps += __shfl_xor(ps, 2, 32);
      ps += __shfl_xor(ps, 4, 32);
      ps += __shfl_xor(ps, 8, 32);
      rowSum[r] = rowSum[r] * corr + ps;
      accO[0][r] *= corr;
      accO[1][r] *= corr;
      accO[2][r] *= corr;
      accO[3][r] *= corr;
      // stash P (f16) row-major into private LDS slab: row = r + 8*laneHi
      _Float16* pr = myLds + (r + laneHi * 8) * 64;
      pr[lmod]      = (_Float16)p0;
      pr[16 + lmod] = (_Float16)p1;
      pr[32 + lmod] = (_Float16)p2;
      pr[48 + lmod] = (_Float16)p3;
    }

    // ---- reload P in A-layout: 4x 16B ds loads (wave-private, dscnt-waited) --
    v16h pa0, pa1;
    {
      const _Float16* prow = myLds + lmod * 64;
      v8h c0 = *(const v8h*)(prow + laneHi * 16);        // m 0..31 block
      v8h c1 = *(const v8h*)(prow + laneHi * 16 + 8);
      v8h c2 = *(const v8h*)(prow + 32 + laneHi * 16);   // m 32..63 block
      v8h c3 = *(const v8h*)(prow + 40 + laneHi * 16);
#pragma unroll
      for (int j = 0; j < 8; ++j) {
        pa0[j] = c0[j]; pa0[8 + j] = c1[j];
        pa1[j] = c2[j]; pa1[8 + j] = c3[j];
      }
    }

    // ---- accO += P V : 4 d-tiles x 2 m-halves; V B-tile is contiguous v16h --
#pragma unroll
    for (int t = 0; t < 4; ++t) {
      const _Float16* Vr = V + (size_t)(t * 16 + lmod) * 2048 + m0 + laneHi * 16;
      v16h bv0 = *(const v16h*)(Vr);       // m0 .. m0+31
      v16h bv1 = *(const v16h*)(Vr + 32);  // m0+32 .. m0+63
      accO[t] = WMMA_F16(pa0, bv0, accO[t]);
      accO[t] = WMMA_F16(pa1, bv1, accO[t]);
    }
  }

  // ---- normalize; store channel-contiguous xw[b][n][c], c = d*4 + h ----
#pragma unroll
  for (int r = 0; r < 8; ++r) {
    const float inv = 1.0f / rowSum[r];
    const int n = nBase + r + laneHi * 8;
#pragma unroll
    for (int t = 0; t < 4; ++t) {
      const int d = t * 16 + lmod;
      xw[((size_t)b * 2048 + n) * 256 + (d * 4 + h)] = (_Float16)(accO[t][r] * inv);
    }
  }
}

// ---------------------------------------------------------------------------
// Kernel 3: output projection.  out[b,o,n] = sum_i Wm[o,i]*x[b,n,i] + bm[o]
// x is f16 channel-contiguous ([B,N,256]) -> B-tiles are contiguous v16h
// loads, exactly like the K operand in attention.  Output f32 [B,256,N].
// ---------------------------------------------------------------------------
__global__ __launch_bounds__(128) void proj_out_kernel(
    const _Float16* __restrict__ X, const float* __restrict__ W,
    const float* __restrict__ bias, float* __restrict__ out) {
  const int lane   = threadIdx.x & 31;
  const int wave   = threadIdx.x >> 5;
  const int laneHi = lane >> 4;
  const int lmod   = lane & 15;
  const int kBase  = laneHi * 8;

  const int b     = blockIdx.z;
  const int oBase = blockIdx.y * 16;
  const int nBase = (blockIdx.x * 4 + wave) * 64;

  const float*    Wrow = W + (size_t)(oBase + lmod) * 256;
  const _Float16* Xb   = X + (size_t)b * 2048 * 256;

  v8f acc[4] = {v8f{0}, v8f{0}, v8f{0}, v8f{0}};

  for (int k0 = 0; k0 < 256; k0 += 32) {
    // A tile: weights (f32 -> f16), 4x float4 loads
    v4f w0 = *(const v4f*)(Wrow + k0 + laneHi * 16);
    v4f w1 = *(const v4f*)(Wrow + k0 + laneHi * 16 + 4);
    v4f w2 = *(const v4f*)(Wrow + k0 + laneHi * 16 + 8);
    v4f w3 = *(const v4f*)(Wrow + k0 + laneHi * 16 + 12);
    v16h a;
#pragma unroll
    for (int j = 0; j < 4; ++j) {
      a[j]      = (_Float16)w0[j];
      a[4 + j]  = (_Float16)w1[j];
      a[8 + j]  = (_Float16)w2[j];
      a[12 + j] = (_Float16)w3[j];
    }
#pragma unroll
    for (int t = 0; t < 4; ++t) {
      const int n = nBase + t * 16 + lmod;
      // B-layout: K = k0 + 16*laneHi + j, contiguous in x[b][n][.]
      v16h bm = *(const v16h*)(Xb + (size_t)n * 256 + k0 + laneHi * 16);
      acc[t] = WMMA_F16(a, bm, acc[t]);
    }
  }

#pragma unroll
  for (int t = 0; t < 4; ++t) {
    const int n = nBase + t * 16 + lmod;
#pragma unroll
    for (int r = 0; r < 8; ++r) {
      const int o = oBase + r + laneHi * 8;
      out[((size_t)b * 256 + o) * 2048 + n] = acc[t][r] + bias[o];
    }
  }
}

// ---------------------------------------------------------------------------
extern "C" void kernel_launch(void* const* d_in, const int* in_sizes, int n_in,
                              void* d_out, int out_size, void* d_ws, size_t ws_size,
                              hipStream_t stream) {
  const float* query = (const float*)d_in[0];
  const float* key   = (const float*)d_in[1];
  const float* value = (const float*)d_in[2];
  const float* Wq    = (const float*)d_in[3];
  const float* bq    = (const float*)d_in[4];
  const float* Wk    = (const float*)d_in[5];
  const float* bk    = (const float*)d_in[6];
  const float* Wv    = (const float*)d_in[7];
  const float* bv    = (const float*)d_in[8];
  const float* Wm    = (const float*)d_in[9];
  const float* bm    = (const float*)d_in[10];
  float* out = (float*)d_out;

  const size_t per = (size_t)8 * 4 * 2048 * 64;  // 4M halves = 8 MB each
  _Float16* q_ws = (_Float16*)d_ws;
  _Float16* k_ws = q_ws + per;
  _Float16* v_ws = k_ws + per;
  _Float16* x_ws = v_ws + per;

  dim3 blk(128, 1, 1);
  dim3 gproj(8, 16, 8);  // n-tiles(256 each), o-tiles(16), batch
  proj_qkv_kernel<<<gproj, blk, 0, stream>>>(query, Wq, bq, q_ws, 0);
  proj_qkv_kernel<<<gproj, blk, 0, stream>>>(key,   Wk, bk, k_ws, 1);
  proj_qkv_kernel<<<gproj, blk, 0, stream>>>(value, Wv, bv, v_ws, 2);

  dim3 gattn(32, 4, 8);  // n-tiles(64 rows/block), heads, batch
  attn_kernel<<<gattn, blk, 0, stream>>>(q_ws, k_ws, v_ws, x_ws);

  proj_out_kernel<<<gproj, blk, 0, stream>>>(x_ws, Wm, bm, out);
}
